// CLF_CBF_QP_Net_34299608826619
// MI455X (gfx1250) — matrix operand order repaired and checked by reading, not compiled, not run on hardware
//
#include <hip/hip_runtime.h>
#include <hip/hip_bf16.h>
#include <math.h>

// ---------------- problem constants (from reference) ----------------
#define HD        256
#define NIN       4
#define TB        16          // batch rows per workgroup
#define NTHREADS  256         // 8 wave32s
#define CLF_LAM   1.0f
#define CBF_LAM   1.0f
#define P_V       10.0f
#define P_H       100.0f

typedef __attribute__((ext_vector_type(16))) __bf16 v16bf;
typedef __attribute__((ext_vector_type(8)))  __bf16 v8bf;
typedef __attribute__((ext_vector_type(8)))  float  v8f;

// =====================================================================
// Prep: cast big weights to bf16 in row-major AND transposed layouts so
// every WMMA B-fragment is a contiguous 32B per-lane read (L2 resident).
// Also build zero-padded [16][256] W1^T matrices for the 16x4 grad GEMMs.
// =====================================================================
__global__ void prep_weights(const float* __restrict__ W1v,
                             const float* __restrict__ W2v,
                             const float* __restrict__ W1h,
                             const float* __restrict__ W2h,
                             __bf16* __restrict__ W2v_bf,  __bf16* __restrict__ W2vT_bf,
                             __bf16* __restrict__ W2h_bf,  __bf16* __restrict__ W2hT_bf,
                             __bf16* __restrict__ W1vT_bf, __bf16* __restrict__ W1hT_bf)
{
    const int h = blockIdx.x;   // 0..255
    const int k = threadIdx.x;  // 0..255
    const float wv = W2v[h * HD + k];
    const float wh = W2h[h * HD + k];
    W2v_bf [h * HD + k] = (__bf16)wv;
    W2vT_bf[k * HD + h] = (__bf16)wv;
    W2h_bf [h * HD + k] = (__bf16)wh;
    W2hT_bf[k * HD + h] = (__bf16)wh;
    if (h < 16) {
        // W1?T_pad[n][k] = (n<4) ? W1?[k][n] : 0
        const float av = (h < NIN) ? W1v[k * NIN + h] : 0.0f;
        const float ah = (h < NIN) ? W1h[k * NIN + h] : 0.0f;
        W1vT_bf[h * HD + k] = (__bf16)av;
        W1hT_bf[h * HD + k] = (__bf16)ah;
    }
}

// ---------------------------------------------------------------------
// One 16x16 output tile, K=256 in 8 chained v_wmma_f32_16x16x32_bf16.
//  A: LDS, row-major [TB][HD] bf16.  A-layout (16-bit 16x32):
//     lane L: M=L%16; halves 0..7 = K kb+base..kb+base+7,
//                     halves 8..15 = K kb+base+8.., base = 16*(L/16).
//  B: global row pointer for this lane's output column.
//     B order used here, matching A (16-bit KxN): lane L holds column N=L%16,
//     K = kb+16*(L/16) .. +15  -> one contiguous 32B chunk per step.
// ---------------------------------------------------------------------
__device__ __forceinline__ v8f gemm_tile_k256(const __bf16* __restrict__ Alds,
                                              const __bf16* __restrict__ Brow,
                                              int lane)
{
    const int M     = lane & 15;
    const int abase = (lane >> 4) << 3;   // 0 or 8
    const int koff  = (lane >> 4) << 4;   // 0 or 16
    v8f acc = {0.f, 0.f, 0.f, 0.f, 0.f, 0.f, 0.f, 0.f};
    const __bf16* arow = Alds + M * HD;
#pragma unroll
    for (int kb = 0; kb < HD; kb += 32) {
        v8bf lo = *(const v8bf*)(arow + kb + koff);
        v8bf hi = *(const v8bf*)(arow + kb + koff + 8);
        v16bf afrag;
#pragma unroll
        for (int i = 0; i < 8; ++i) { afrag[i] = lo[i]; afrag[8 + i] = hi[i]; }
        v16bf bfrag = *(const v16bf*)(Brow + kb + koff);
        acc = __builtin_amdgcn_wmma_f32_16x16x32_bf16(false, afrag, false, bfrag,
                                                      (short)0, acc, false, false);
    }
    return acc;
}

// =====================================================================
// Fused kernel: 16 batch rows / workgroup, 8 waves, Hd=256 columns split
// as 2 column tiles per wave.  All accumulation + nonlinear math in f32.
// =====================================================================
__global__ __launch_bounds__(NTHREADS)
void clf_cbf_qp_kernel(const float* __restrict__ x,
                       const float* __restrict__ W1v, const float* __restrict__ b1v,
                       const float* __restrict__ b2v,
                       const float* __restrict__ W1h, const float* __restrict__ b1h,
                       const float* __restrict__ b2h,
                       const float* __restrict__ W3h, const float* __restrict__ b3h,
                       const float* __restrict__ A_dyn, const float* __restrict__ B_dyn,
                       const __bf16* __restrict__ W2v_bf,  const __bf16* __restrict__ W2vT_bf,
                       const __bf16* __restrict__ W2h_bf,  const __bf16* __restrict__ W2hT_bf,
                       const __bf16* __restrict__ W1vT_bf, const __bf16* __restrict__ W1hT_bf,
                       float* __restrict__ out, int Btot)
{
    __shared__ float  xs [TB][NIN];
    __shared__ __bf16 v1b[TB][HD];    // tanh(layer1_v) as WMMA-A operand
    __shared__ __bf16 h1b[TB][HD];
    __shared__ float  dv1[TB][HD];    // 1 - v1^2   (kept f32)
    __shared__ float  dh1[TB][HD];
    __shared__ __bf16 ab [TB][HD];    // a  = v2(1-v2^2)
    __shared__ __bf16 bb [TB][HD];    // b' = W3h*(1-h2^2)
    __shared__ __bf16 tVb[TB][HD];    // t  = (a@W2v) * dv1
    __shared__ __bf16 tHb[TB][HD];
    __shared__ float  Vacc[TB];       // 0.5*sum v2^2
    __shared__ float  Hacc[TB];       // sum W3h*h2  (pre-tanh H)
    __shared__ float  gV[TB][NIN];
    __shared__ float  gH[TB][NIN];

    const int tid  = threadIdx.x;
    const int lane = tid & 31;
    const int wave = tid >> 5;
    const int n    = lane & 15;            // C-frag column within tile
    const int mb   = (lane >> 4) << 3;     // C-frag row base (0 or 8)
    const int b0   = blockIdx.x * TB;

    // ---- phase 0: stage x tile, zero reductions ----
    if (tid < TB * NIN) xs[tid >> 2][tid & 3] = x[(b0 + (tid >> 2)) * NIN + (tid & 3)];
    if (tid < TB) { Vacc[tid] = 0.f; Hacc[tid] = 0.f; }
    __syncthreads();

    // ---- phase 1: layer 1 (K=4 -> pure VALU), thread = hidden column ----
    {
        const int h = tid;
        const float w0 = W1v[h*4+0], w1 = W1v[h*4+1], w2 = W1v[h*4+2], w3 = W1v[h*4+3];
        const float u0 = W1h[h*4+0], u1 = W1h[h*4+1], u2 = W1h[h*4+2], u3 = W1h[h*4+3];
        const float bv = b1v[h], bh = b1h[h];
#pragma unroll
        for (int b = 0; b < TB; ++b) {
            float pv = bv + xs[b][0]*w0 + xs[b][1]*w1 + xs[b][2]*w2 + xs[b][3]*w3;
            float v  = tanhf(pv);
            v1b[b][h] = (__bf16)v;
            dv1[b][h] = 1.0f - v * v;
            float ph = bh + xs[b][0]*u0 + xs[b][1]*u1 + xs[b][2]*u2 + xs[b][3]*u3;
            float hv = tanhf(ph);
            h1b[b][h] = (__bf16)hv;
            dh1[b][h] = 1.0f - hv * hv;
        }
    }
    __syncthreads();

    // ---- phase 2: layer-2 GEMMs (WMMA) + fused epilogues ----
#pragma unroll
    for (int t = 0; t < 2; ++t) {
        const int cb = (wave * 2 + t) * 16;
        const int h  = cb + n;
        // v-side: v2 = tanh(v1 @ W2v^T + b2v);  B[k][n]=W2v[h][k] -> row h of W2v
        v8f cv = gemm_tile_k256(&v1b[0][0], W2v_bf + h * HD, lane);
        const float biasv = b2v[h];
#pragma unroll
        for (int r = 0; r < 8; ++r) {
            const int m = mb + r;
            float v2 = tanhf(cv[r] + biasv);
            ab[m][h] = (__bf16)(v2 * (1.0f - v2 * v2));
            atomicAdd(&Vacc[m], 0.5f * v2 * v2);
        }
        // h-side
        v8f ch = gemm_tile_k256(&h1b[0][0], W2h_bf + h * HD, lane);
        const float biash = b2h[h];
        const float w3    = W3h[h];
#pragma unroll
        for (int r = 0; r < 8; ++r) {
            const int m = mb + r;
            float h2 = tanhf(ch[r] + biash);
            bb[m][h] = (__bf16)(w3 * (1.0f - h2 * h2));
            atomicAdd(&Hacc[m], w3 * h2);
        }
    }
    __syncthreads();

    // ---- phase 3: c = a @ W2v, d = b' @ W2h; t = c*dv1 (WMMA) ----
#pragma unroll
    for (int t = 0; t < 2; ++t) {
        const int cb = (wave * 2 + t) * 16;
        const int kc = cb + n;   // output k-column
        // B[K=h][N=k] = W2v[h][k] -> contiguous h = row kc of W2v^T
        v8f cv = gemm_tile_k256(&ab[0][0], W2vT_bf + kc * HD, lane);
        v8f ch = gemm_tile_k256(&bb[0][0], W2hT_bf + kc * HD, lane);
#pragma unroll
        for (int r = 0; r < 8; ++r) {
            const int m = mb + r;
            tVb[m][kc] = (__bf16)(cv[r] * dv1[m][kc]);
            tHb[m][kc] = (__bf16)(ch[r] * dh1[m][kc]);
        }
    }
    __syncthreads();

    // ---- phase 4: grad = t @ W1 (16x4 via zero-padded 16-col WMMA) ----
    if (wave == 0) {
        v8f g = gemm_tile_k256(&tVb[0][0], W1vT_bf + n * HD, lane);
        if (n < NIN) {
#pragma unroll
            for (int r = 0; r < 8; ++r) gV[mb + r][n] = g[r];
        }
    } else if (wave == 1) {
        v8f g = gemm_tile_k256(&tHb[0][0], W1hT_bf + n * HD, lane);
        if (n < NIN) {
#pragma unroll
            for (int r = 0; r < 8; ++r) gH[mb + r][n] = g[r];
        }
    }
    __syncthreads();

    // ---- phase 5: closed-form QP + outputs (one thread per batch row) ----
    if (tid < TB) {
        const int b = tid;
        const float V  = Vacc[b];
        const float H  = tanhf(Hacc[b] + b3h[0]);
        const float dH = 1.0f - H * H;
        float gv[NIN], gh[NIN], f[NIN];
#pragma unroll
        for (int i = 0; i < NIN; ++i) { gv[i] = gV[b][i]; gh[i] = dH * gH[b][i]; }
#pragma unroll
        for (int i = 0; i < NIN; ++i) {   // f = x @ A_dyn^T
            f[i] = xs[b][0] * A_dyn[i*4+0] + xs[b][1] * A_dyn[i*4+1]
                 + xs[b][2] * A_dyn[i*4+2] + xs[b][3] * A_dyn[i*4+3];
        }
        float LfV = 0.f, LfH = 0.f, g1 = 0.f, g2 = 0.f;
#pragma unroll
        for (int i = 0; i < NIN; ++i) {
            LfV += gv[i] * f[i];  LfH += gh[i] * f[i];
            g1  += gv[i] * B_dyn[i];  g2 += gh[i] * B_dyn[i];
        }
        const float q1 = LfV + CLF_LAM * V;
        const float q2 = LfH + CBF_LAM * H;

        // 4 KKT candidates, pick first feasible (argmax-of-bool semantics)
        const float s1v[4] = {0.f, 0.f, 1.f, 1.f};
        const float s2v[4] = {0.f, 1.f, 0.f, 1.f};
        float u = 0.f; bool found = false;
#pragma unroll
        for (int i = 0; i < 4; ++i) {
            const float s1 = s1v[i], s2 = s2v[i];
            const float den = 1.0f + 2.0f*P_V*s1*g1*g1 + 2.0f*P_H*s2*g2*g2;
            const float uu  = -(2.0f*P_V*s1*g1*q1 + 2.0f*P_H*s2*g2*q2) / den;
            const float a1 = q1 + g1 * uu;
            const float a2 = q2 + g2 * uu;
            const bool ok1 = (s1 > 0.f) ? (a1 >= 0.f) : (a1 <= 0.f);
            const bool ok2 = (s2 > 0.f) ? (a2 <= 0.f) : (a2 >= 0.f);
            if (!found && ok1 && ok2) { u = uu; found = true; }
        }
        const float rV   = fmaxf(q1 + g1 * u, 0.f);
        const float rH   = fmaxf(-(q2 + g2 * u), 0.f);
        const float Vdot = LfV + g1 * u;
        const float Hdot = LfH + g2 * u;
        const int gb = b0 + b;
        out[0 * Btot + gb] = u;
        out[1 * Btot + gb] = rV;
        out[2 * Btot + gb] = rH;
        out[3 * Btot + gb] = V;
        out[4 * Btot + gb] = Vdot;
        out[5 * Btot + gb] = H;
        out[6 * Btot + gb] = Hdot;
    }
}

// =====================================================================
extern "C" void kernel_launch(void* const* d_in, const int* in_sizes, int n_in,
                              void* d_out, int out_size, void* d_ws, size_t ws_size,
                              hipStream_t stream)
{
    const float* x     = (const float*)d_in[0];
    const float* W1v   = (const float*)d_in[1];
    const float* b1v   = (const float*)d_in[2];
    const float* W2v   = (const float*)d_in[3];
    const float* b2v   = (const float*)d_in[4];
    const float* W1h   = (const float*)d_in[5];
    const float* b1h   = (const float*)d_in[6];
    const float* W2h   = (const float*)d_in[7];
    const float* b2h   = (const float*)d_in[8];
    const float* W3h   = (const float*)d_in[9];
    const float* b3h   = (const float*)d_in[10];
    const float* A_dyn = (const float*)d_in[11];
    const float* B_dyn = (const float*)d_in[12];
    float* out = (float*)d_out;
    const int Btot = in_sizes[0] / NIN;    // 65536

    // workspace carve-up (bf16 weight copies), 32B-aligned chunks
    char* ws = (char*)d_ws;
    const size_t SZ2 = (size_t)HD * HD * sizeof(__bf16);   // 128 KB
    const size_t SZ1 = (size_t)16 * HD * sizeof(__bf16);   // 8 KB
    __bf16* W2v_bf  = (__bf16*)(ws);
    __bf16* W2vT_bf = (__bf16*)(ws + SZ2);
    __bf16* W2h_bf  = (__bf16*)(ws + 2 * SZ2);
    __bf16* W2hT_bf = (__bf16*)(ws + 3 * SZ2);
    __bf16* W1vT_bf = (__bf16*)(ws + 4 * SZ2);
    __bf16* W1hT_bf = (__bf16*)(ws + 4 * SZ2 + SZ1);

    prep_weights<<<HD, HD, 0, stream>>>(W1v, W2v, W1h, W2h,
                                        W2v_bf, W2vT_bf, W2h_bf, W2hT_bf,
                                        W1vT_bf, W1hT_bf);

    const int blocks = Btot / TB;          // 4096
    clf_cbf_qp_kernel<<<blocks, NTHREADS, 0, stream>>>(
        x, W1v, b1v, b2v, W1h, b1h, b2h, W3h, b3h, A_dyn, B_dyn,
        W2v_bf, W2vT_bf, W2h_bf, W2hT_bf, W1vT_bf, W1hT_bf,
        out, Btot);
}
